// Stage_31464930411045
// MI455X (gfx1250) — matrix operand order repaired and checked by reading, not compiled, not run on hardware
//
#include <hip/hip_runtime.h>
#include <hip/hip_bf16.h>

#define N_PTS   16384
#define DM      256
#define KNB     27
#define GSZ     4096
#define ORDER_  3
#define LSEQ    12288          /* ORDER_*GSZ tokens per batch sequence   */
#define LTOT    49152          /* 4 * LSEQ total tokens                  */
#define DI      512
#define KCONV   6912           /* KNB * DM                               */

typedef __attribute__((ext_vector_type(16))) _Float16 v16h;
typedef __attribute__((ext_vector_type(8)))  _Float16 v8h;
typedef __attribute__((ext_vector_type(8)))  float    v8f;
typedef __attribute__((ext_vector_type(4)))  unsigned int u32x4;
typedef __attribute__((ext_vector_type(8)))  int      i32x8;
typedef __attribute__((ext_vector_type(4)))  int      i32x4;

#define CEILDIV(a,b) (((a)+(b)-1)/(b))

static __device__ __forceinline__ float siluf(float x) { return x / (1.f + __expf(-x)); }

// ---------------------------------------------------------------------------
// Tensor Data Mover: DMA a 2D tile (64 rows x 32 f16) of a row-major f16
// matrix (row stride = stride_elems) into LDS at lds_off. Rows >= rows_avail
// are zero-filled by TDM OOB handling. data_size = 4B units.
// ---------------------------------------------------------------------------
static __device__ __forceinline__ void
tdm_load_tile(unsigned lds_off, const _Float16* gptr, int rows_avail, int stride_elems)
{
    unsigned long long ga = (unsigned long long)gptr;
    const unsigned td0 = (unsigned)(stride_elems >> 1);   // row length, u32 units
    const unsigned td1 = (unsigned)rows_avail;            // valid rows (zero-fill past)
    const unsigned long long s0 = (unsigned long long)(stride_elems >> 1);

    u32x4 g0;
    g0[0] = 1u;                                           // count=1, user descriptor
    g0[1] = lds_off;                                      // lds_addr (bytes)
    g0[2] = (unsigned)(ga & 0xffffffffu);                 // global_addr[31:0]
    g0[3] = (unsigned)((ga >> 32) & 0x01ffffffu) | (2u << 30);  // addr[56:32] | type=2

    i32x8 g1;
    g1[0] = (int)(2u << 16);                              // workgroup_mask=0, data_size=4B
    g1[1] = (int)((td0 & 0xffffu) << 16);                 // tensor_dim0[15:0]
    g1[2] = (int)((td0 >> 16) | ((td1 & 0xffffu) << 16)); // tensor_dim0[31:16] | dim1[15:0]
    g1[3] = (int)((td1 >> 16) | (16u << 16));             // dim1[31:16] | tile_dim0=16 u32
    g1[4] = (int)64;                                      // tile_dim1=64 rows, tile_dim2=0
    g1[5] = (int)(s0 & 0xffffffffu);                      // dim0_stride[31:0]
    g1[6] = (int)((s0 >> 32) & 0xffffu);                  // dim0_stride[47:32]
    g1[7] = 0;

    i32x4 gz = {0, 0, 0, 0};
#if defined(__clang_major__) && (__clang_major__ >= 23)
    i32x8 gz8 = {0, 0, 0, 0, 0, 0, 0, 0};
    __builtin_amdgcn_tensor_load_to_lds(g0, g1, gz, gz, gz8, 0);
#else
    __builtin_amdgcn_tensor_load_to_lds(g0, g1, gz, gz, 0);
#endif
}

// ---------------------------------------------------------------------------
// WMMA GEMM with TDM-staged B tiles:
//   out[M,Nc] = act( A16[M,Kc] * W16[Nc,Kc]^T + bias )
// Block: 256 threads = 8 waves. Block tile 128(M) x 64(N); wave w owns rows
// [m0, m0+16) and 4 accumulators covering 64 cols (A fragment reused 4x).
// B tile (64 x 32 f16) double-buffered in LDS via tensor_load_to_lds issued
// by wave 0, synchronized with s_wait_tensorcnt + block barrier.
// act: 0 none, 1 silu, 2 softplus
// ---------------------------------------------------------------------------
__global__ void __launch_bounds__(256)
wmma_gemm(const _Float16* __restrict__ A, const _Float16* __restrict__ W,
          const float* __restrict__ bias, float* __restrict__ out32,
          _Float16* __restrict__ out16, int M, int Nc, int Kc, int act)
{
    __shared__ __align__(32) _Float16 sW[2][64 * 32];
    const int wid  = threadIdx.x >> 5;
    const int lane = threadIdx.x & 31;
    const int half = lane >> 4;
    const int l16  = lane & 15;
    const int m0   = blockIdx.y * 128 + wid * 16;
    const int n0b  = blockIdx.x * 64;
    const int rowsW = Nc - n0b;

    const _Float16* pA = A + (size_t)(m0 + l16) * Kc;
    const _Float16* gW = W + (size_t)n0b * Kc;

    v8f acc[4] = {v8f{}, v8f{}, v8f{}, v8f{}};

    if (threadIdx.x == 0)
        tdm_load_tile((unsigned)(unsigned long long)&sW[0][0], gW, rowsW, Kc);

    int cur = 0;
    for (int k0 = 0; k0 < Kc; k0 += 32) {
        if (threadIdx.x == 0) {
            if (k0 + 32 < Kc) {
                tdm_load_tile((unsigned)(unsigned long long)&sW[cur ^ 1][0],
                              gW + k0 + 32, rowsW, Kc);
                __builtin_amdgcn_s_wait_tensorcnt(1);   // current tile landed
            } else {
                __builtin_amdgcn_s_wait_tensorcnt(0);
            }
        }
        __syncthreads();

        v8h a0 = *(const v8h*)(pA + k0 + half * 16);
        v8h a1 = *(const v8h*)(pA + k0 + half * 16 + 8);
        if (k0 + 32 < Kc) __builtin_prefetch(pA + k0 + 32, 0, 3);
        v16h a;
#pragma unroll
        for (int e = 0; e < 8; ++e) { a[e] = a0[e]; a[8 + e] = a1[e]; }

        const _Float16* sb = &sW[cur][0];
#pragma unroll
        for (int nt = 0; nt < 4; ++nt) {
            v16h b = *(const v16h*)(sb + (nt * 16 + l16) * 32 + half * 16);
            acc[nt] = __builtin_amdgcn_wmma_f32_16x16x32_f16(false, a, false, b,
                                                             (short)0, acc[nt],
                                                             false, false);
        }
        __syncthreads();
        cur ^= 1;
    }

#pragma unroll
    for (int nt = 0; nt < 4; ++nt) {
        const int col = n0b + nt * 16 + l16;
        if (col < Nc) {
            const float bb = bias ? bias[col] : 0.f;
#pragma unroll
            for (int r = 0; r < 8; ++r) {
                const int row = m0 + half * 8 + r;
                float v = acc[nt][r] + bb;
                if (act == 1)      v = siluf(v);
                else if (act == 2) v = (v > 20.f) ? v : log1pf(__expf(v));
                if (out32) out32[(size_t)row * Nc + col] = v;
                if (out16) out16[(size_t)row * Nc + col] = (_Float16)v;
            }
        }
    }
}

// ---------------------------------------------------------------------------
// Gathered-A WMMA GEMM for the neighbor conv (A rows are indirect, so this
// one keeps per-wave global loads + prefetch):
// out[n,d] = sum_{k,c} feat[nidx[n,k],c] * convW^T[d, k*256+c] + bias[d]
// ---------------------------------------------------------------------------
__global__ void __launch_bounds__(256)
wmma_gather_conv(const _Float16* __restrict__ F, const int* __restrict__ nidx,
                 const _Float16* __restrict__ W, const float* __restrict__ bias,
                 _Float16* __restrict__ out16)
{
    const int wid  = threadIdx.x >> 5;
    const int lane = threadIdx.x & 31;
    const int half = lane >> 4;
    const int l16  = lane & 15;
    const int m0 = blockIdx.y * 32 + (wid >> 2) * 16;
    const int n0 = blockIdx.x * 64 + (wid & 3) * 16;
    const int rowA = m0 + l16;
    const _Float16* pW = W + (size_t)(n0 + l16) * KCONV;

    v8f c = {};
    for (int kn = 0; kn < KNB; ++kn) {
        const int idx = nidx[rowA * KNB + kn];
        const _Float16* pF = F + (size_t)idx * DM;
        for (int cb = 0; cb < DM; cb += 32) {
            const int k0 = kn * DM + cb;
            v8h a0 = *(const v8h*)(pF + cb + half * 16);
            v8h a1 = *(const v8h*)(pF + cb + half * 16 + 8);
            v16h b = *(const v16h*)(pW + k0 + half * 16);
            v16h a;
#pragma unroll
            for (int e = 0; e < 8; ++e) { a[e] = a0[e]; a[8 + e] = a1[e]; }
            __builtin_prefetch(pW + k0 + 32, 0, 3);
            c = __builtin_amdgcn_wmma_f32_16x16x32_f16(false, a, false, b,
                                                       (short)0, c, false, false);
        }
    }
    const int col = n0 + l16;
#pragma unroll
    for (int r = 0; r < 8; ++r) {
        int row = m0 + half * 8 + r;
        out16[(size_t)row * DM + col] = (_Float16)(c[r] + bias[col]);
    }
}

// ---------------------------------------------------------------------------
// LayerNorm over 256 cols. One block (256 threads) per row.
// y = LN(in)*s + b (+ padd)   ; act 1 = exact gelu
// ---------------------------------------------------------------------------
__global__ void __launch_bounds__(256)
ln256(const float* __restrict__ in, const float* __restrict__ s,
      const float* __restrict__ b, const float* __restrict__ padd,
      float* __restrict__ out32, _Float16* __restrict__ out16, int act)
{
    __shared__ float rs[256], rq[256];
    const int row = blockIdx.x, c = threadIdx.x;
    float x = in[(size_t)row * 256 + c];
    rs[c] = x; rq[c] = x * x;
    __syncthreads();
    for (int off = 128; off > 0; off >>= 1) {
        if (c < off) { rs[c] += rs[c + off]; rq[c] += rq[c + off]; }
        __syncthreads();
    }
    const float m   = rs[0] * (1.f / 256.f);
    const float var = rq[0] * (1.f / 256.f) - m * m;
    float y = (x - m) * rsqrtf(var + 1e-5f) * s[c] + b[c];
    if (padd) y += padd[(size_t)row * 256 + c];
    if (act == 1) y = 0.5f * y * (1.f + erff(y * 0.70710678118f));
    if (out32) out32[(size_t)row * 256 + c] = y;
    if (out16) out16[(size_t)row * 256 + c] = (_Float16)y;
}

// ---------------------------------------------------------------------------
// Depthwise causal conv1d (width 4) + silu over the [LTOT,1024] xz buffer.
// ---------------------------------------------------------------------------
__global__ void conv1d_silu(const _Float16* __restrict__ xz,
                            const float* __restrict__ w, const float* __restrict__ bv,
                            float* __restrict__ xc, _Float16* __restrict__ xc16)
{
    long long id = (long long)blockIdx.x * 256 + threadIdx.x;
    if (id >= (long long)LTOT * DI) return;
    const int e = (int)(id & (DI - 1));
    const long long t = id >> 9;
    const int p = (int)(t % LSEQ);
    float acc = bv[e];
    const float w0 = w[e * 4], w1 = w[e * 4 + 1], w2 = w[e * 4 + 2], w3 = w[e * 4 + 3];
    if (p >= 3) acc += w0 * (float)xz[(t - 3) * 1024 + e];
    if (p >= 2) acc += w1 * (float)xz[(t - 2) * 1024 + e];
    if (p >= 1) acc += w2 * (float)xz[(t - 1) * 1024 + e];
    acc += w3 * (float)xz[t * 1024 + e];
    const float sv = siluf(acc);
    xc[id] = sv; xc16[id] = (_Float16)sv;
}

// ---------------------------------------------------------------------------
// Selective scan. 4 blocks (one per batch sequence) x 512 threads (one per
// inner channel). Per-step B/C broadcast via LDS.
// ---------------------------------------------------------------------------
__global__ void __launch_bounds__(512)
mamba_scan(const float* __restrict__ dbc, const float* __restrict__ dt,
           const float* __restrict__ xc, const _Float16* __restrict__ xz,
           const float* __restrict__ A_log, const float* __restrict__ Dsk,
           float* __restrict__ y, _Float16* __restrict__ y16)
{
    const int b = blockIdx.x, e = threadIdx.x;
    __shared__ float bc[32];
    float a[16], h[16];
#pragma unroll
    for (int i = 0; i < 16; ++i) { a[i] = -__expf(A_log[e * 16 + i]); h[i] = 0.f; }
    const float dsk = Dsk[e];
    for (int p = 0; p < LSEQ; ++p) {
        const long long t = (long long)b * LSEQ + p;
        __syncthreads();
        if (e < 32) bc[e] = dbc[t * 48 + 16 + e];
        __syncthreads();
        const float xt  = xc[t * DI + e];
        const float dtt = dt[t * DI + e];
        float acc = 0.f;
#pragma unroll
        for (int i = 0; i < 16; ++i) {
            h[i] = h[i] * __expf(dtt * a[i]) + (dtt * xt) * bc[i];
            acc += h[i] * bc[16 + i];
        }
        const float zv = (float)xz[t * 1024 + DI + e];
        const float yv = (acc + xt * dsk) * siluf(zv);
        y[t * DI + e] = yv; y16[t * DI + e] = (_Float16)yv;
    }
}

// ------------------------- small elementwise helpers -----------------------
__global__ void cvt_f16(const float* __restrict__ in, _Float16* __restrict__ o, long long n)
{
    long long id = (long long)blockIdx.x * 256 + threadIdx.x;
    if (id < n) o[id] = (_Float16)in[id];
}
__global__ void cvt_convw(const float* __restrict__ cw, _Float16* __restrict__ o)
{   // o[d*6912+kk] = cw[kk*256+d]
    long long id = (long long)blockIdx.x * 256 + threadIdx.x;
    if (id >= (long long)DM * KCONV) return;
    const int kk = (int)(id % KCONV), d = (int)(id / KCONV);
    o[id] = (_Float16)cw[(long long)kk * DM + d];
}
__global__ void cvt_dtpadw(const float* __restrict__ w, _Float16* __restrict__ o)
{   // [2*512, 32]: cols 0..15 = dt_proj_w row, 16..31 = 0
    long long id = (long long)blockIdx.x * 256 + threadIdx.x;
    if (id >= 2LL * DI * 32) return;
    const int r = (int)(id & 31); const long long er = id >> 5;
    o[id] = (r < 16) ? (_Float16)w[er * 16 + r] : (_Float16)0.f;
}
__global__ void build_dtpad(const float* __restrict__ dbc, _Float16* __restrict__ o)
{   // [LTOT,32] padded dt-rank activations
    long long id = (long long)blockIdx.x * 256 + threadIdx.x;
    if (id >= (long long)LTOT * 32) return;
    const int r = (int)(id & 31); const long long t = id >> 5;
    o[id] = (r < 16) ? (_Float16)dbc[t * 48 + r] : (_Float16)0.f;
}
__global__ void build_seq(const float* __restrict__ x, const int* __restrict__ inv,
                          float* __restrict__ out)
{   // out[(b, o*G+g), d] = x[inverse[o, b*G+g], d]
    long long id = (long long)blockIdx.x * 256 + threadIdx.x;
    if (id >= (long long)LTOT * DM) return;
    const int d = (int)(id & 255); const long long t = id >> 8;
    const int bb = (int)(t / LSEQ), r = (int)(t % LSEQ);
    const int o = r >> 12, g = r & 4095;
    const int src = inv[o * N_PTS + bb * GSZ + g];
    out[id] = x[(size_t)src * DM + d];
}
__global__ void build_h3(const float* __restrict__ hsF, const int* __restrict__ ord,
                         _Float16* __restrict__ h3)
{   // h3[j, o*256+d] = hsF[token(order[o,j], o), d]
    long long id = (long long)blockIdx.x * 256 + threadIdx.x;
    if (id >= (long long)N_PTS * 768) return;
    const int c = (int)(id % 768); const long long j = id / 768;
    const int o = c >> 8, d = c & 255;
    const int i = ord[o * N_PTS + (int)j];
    const long long t = (long long)(i >> 12) * LSEQ + o * GSZ + (i & 4095);
    h3[id] = (_Float16)hsF[t * DM + d];
}
__global__ void add_inplace(float* __restrict__ a, const float* __restrict__ b, long long n)
{
    long long id = (long long)blockIdx.x * 256 + threadIdx.x;
    if (id < n) a[id] += b[id];
}

// ---------------------------------------------------------------------------
extern "C" void kernel_launch(void* const* d_in, const int* in_sizes, int n_in,
                              void* d_out, int out_size, void* d_ws, size_t ws_size,
                              hipStream_t stream)
{
    const float* feat      = (const float*)d_in[0];
    const int*   nidx      = (const int*)d_in[1];
    const int*   order     = (const int*)d_in[2];
    const int*   inverse   = (const int*)d_in[3];
    const float* conv_w    = (const float*)d_in[5];
    const float* conv_b    = (const float*)d_in[6];
    const float* cpe_fc_w  = (const float*)d_in[7];
    const float* cpe_fc_b  = (const float*)d_in[8];
    const float* cpe_ln_s  = (const float*)d_in[9];
    const float* cpe_ln_b  = (const float*)d_in[10];
    const float* in_proj_w = (const float*)d_in[11];
    const float* conv1d_w  = (const float*)d_in[12];
    const float* conv1d_b  = (const float*)d_in[13];
    const float* x_proj_w  = (const float*)d_in[14];
    const float* dt_proj_w = (const float*)d_in[15];
    const float* dt_proj_b = (const float*)d_in[16];
    const float* A_log     = (const float*)d_in[17];
    const float* D_skip    = (const float*)d_in[18];
    const float* out_proj_w= (const float*)d_in[19];
    const float* blk_ln_s  = (const float*)d_in[20];
    const float* blk_ln_b  = (const float*)d_in[21];
    const float* normf_s   = (const float*)d_in[22];
    const float* normf_b   = (const float*)d_in[23];
    const float* fuse_w1   = (const float*)d_in[24];
    const float* fuse_b1   = (const float*)d_in[25];
    const float* fuse_ln_s = (const float*)d_in[26];
    const float* fuse_ln_b = (const float*)d_in[27];
    const float* fuse_w2   = (const float*)d_in[28];
    const float* fuse_b2   = (const float*)d_in[29];

    char* base = (char*)d_ws; size_t off = 0;
    auto alloc = [&](size_t bytes) -> void* {
        void* p = base + off; off = (off + bytes + 255) & ~(size_t)255; return p;
    };
    _Float16* featf16  = (_Float16*)alloc(2LL * N_PTS * DM);
    _Float16* cwT16    = (_Float16*)alloc(2LL * DM * KCONV);
    _Float16* cpe16    = (_Float16*)alloc(2LL * DM * DM);
    _Float16* inproj16 = (_Float16*)alloc(2LL * 2 * 1024 * DM);
    _Float16* xproj16  = (_Float16*)alloc(2LL * 2 * 48 * DI);
    _Float16* dtp16    = (_Float16*)alloc(2LL * 2 * DI * 32);
    _Float16* outp16   = (_Float16*)alloc(2LL * 2 * DM * DI);
    _Float16* f1w16    = (_Float16*)alloc(2LL * DM * 768);
    _Float16* f2w16    = (_Float16*)alloc(2LL * DM * DM);
    _Float16* t0h      = (_Float16*)alloc(2LL * N_PTS * DM);
    float*    tmp      = (float*)   alloc(4LL * N_PTS * DM);
    float*    xcpe     = (float*)   alloc(4LL * N_PTS * DM);
    float*    resid    = (float*)   alloc(4LL * LTOT * DM);
    float*    hs       = (float*)   alloc(4LL * LTOT * DM);
    _Float16* hs16     = (_Float16*)alloc(2LL * LTOT * DM);
    _Float16* xz16     = (_Float16*)alloc(2LL * LTOT * 1024);
    float*    xc       = (float*)   alloc(4LL * LTOT * DI);
    _Float16* xc16     = (_Float16*)alloc(2LL * LTOT * DI);
    float*    dbc      = (float*)   alloc(4LL * LTOT * 48);
    _Float16* dtpad16  = (_Float16*)alloc(2LL * LTOT * 32);
    float*    dt       = (float*)   alloc(4LL * LTOT * DI);
    float*    ybuf     = (float*)   alloc(4LL * LTOT * DI);
    _Float16* y16      = (_Float16*)alloc(2LL * LTOT * DI);
    _Float16* h3_16    = (_Float16*)alloc(2LL * N_PTS * 768);
    float*    fu       = (float*)   alloc(4LL * N_PTS * DM);
    _Float16* g16      = (_Float16*)alloc(2LL * N_PTS * DM);

    auto ew = [&](long long n) { return dim3((unsigned)CEILDIV(n, 256)); };

    // ---- weight / input conversion ----
    cvt_f16<<<ew((long long)N_PTS * DM), 256, 0, stream>>>(feat, featf16, (long long)N_PTS * DM);
    cvt_convw<<<ew((long long)DM * KCONV), 256, 0, stream>>>(conv_w, cwT16);
    cvt_f16<<<ew(DM * DM), 256, 0, stream>>>(cpe_fc_w, cpe16, DM * DM);
    cvt_f16<<<ew(2LL * 1024 * DM), 256, 0, stream>>>(in_proj_w, inproj16, 2LL * 1024 * DM);
    cvt_f16<<<ew(2LL * 48 * DI), 256, 0, stream>>>(x_proj_w, xproj16, 2LL * 48 * DI);
    cvt_dtpadw<<<ew(2LL * DI * 32), 256, 0, stream>>>(dt_proj_w, dtp16);
    cvt_f16<<<ew(2LL * DM * DI), 256, 0, stream>>>(out_proj_w, outp16, 2LL * DM * DI);
    cvt_f16<<<ew((long long)DM * 768), 256, 0, stream>>>(fuse_w1, f1w16, (long long)DM * 768);
    cvt_f16<<<ew(DM * DM), 256, 0, stream>>>(fuse_w2, f2w16, DM * DM);

    // ---- CPE: neighbor conv GEMM -> fc -> LN + residual ----
    wmma_gather_conv<<<dim3(DM / 64, N_PTS / 32), 256, 0, stream>>>(
        featf16, nidx, cwT16, conv_b, t0h);
    wmma_gemm<<<dim3(DM / 64, N_PTS / 128), 256, 0, stream>>>(
        t0h, cpe16, cpe_fc_b, tmp, nullptr, N_PTS, DM, DM, 0);
    ln256<<<N_PTS, 256, 0, stream>>>(tmp, cpe_ln_s, cpe_ln_b, feat, xcpe, nullptr, 0);

    // ---- serialize: resid = permuted sequence ----
    build_seq<<<ew((long long)LTOT * DM), 256, 0, stream>>>(xcpe, inverse, resid);

    // ---- 2 Mamba layers ----
    for (int l = 0; l < 2; ++l) {
        if (l > 0)
            add_inplace<<<ew((long long)LTOT * DM), 256, 0, stream>>>(
                resid, hs, (long long)LTOT * DM);
        ln256<<<LTOT, 256, 0, stream>>>(resid, blk_ln_s + l * DM, blk_ln_b + l * DM,
                                        nullptr, hs, hs16, 0);
        // in_proj: [LTOT,256] x [1024,256]^T -> xz16
        wmma_gemm<<<dim3(1024 / 64, LTOT / 128), 256, 0, stream>>>(
            hs16, inproj16 + (size_t)l * 1024 * DM, nullptr, nullptr, xz16,
            LTOT, 1024, DM, 0);
        conv1d_silu<<<ew((long long)LTOT * DI), 256, 0, stream>>>(
            xz16, conv1d_w + l * DI * 4, conv1d_b + l * DI, xc, xc16);
        // x_proj: [LTOT,512] x [48,512]^T -> dbc (TDM zero-fills rows 48..63)
        wmma_gemm<<<dim3(1, LTOT / 128), 256, 0, stream>>>(
            xc16, xproj16 + (size_t)l * 48 * DI, nullptr, dbc, nullptr,
            LTOT, 48, DI, 0);
        build_dtpad<<<ew((long long)LTOT * 32), 256, 0, stream>>>(dbc, dtpad16);
        // dt_proj (+bias, softplus): [LTOT,32] x [512,32]^T -> dt
        wmma_gemm<<<dim3(DI / 64, LTOT / 128), 256, 0, stream>>>(
            dtpad16, dtp16 + (size_t)l * DI * 32, dt_proj_b + l * DI, dt, nullptr,
            LTOT, DI, 32, 2);
        mamba_scan<<<dim3(4), 512, 0, stream>>>(
            dbc, dt, xc, xz16, A_log + (size_t)l * DI * 16, D_skip + l * DI,
            ybuf, y16);
        // out_proj: [LTOT,512] x [256,512]^T -> hs
        wmma_gemm<<<dim3(DM / 64, LTOT / 128), 256, 0, stream>>>(
            y16, outp16 + (size_t)l * DM * DI, nullptr, hs, nullptr,
            LTOT, DM, DI, 0);
    }

    // ---- final residual + LN, de-permute, fuse MLP ----
    add_inplace<<<ew((long long)LTOT * DM), 256, 0, stream>>>(resid, hs, (long long)LTOT * DM);
    ln256<<<LTOT, 256, 0, stream>>>(resid, normf_s, normf_b, nullptr, hs, nullptr, 0);
    build_h3<<<ew((long long)N_PTS * 768), 256, 0, stream>>>(hs, order, h3_16);
    wmma_gemm<<<dim3(DM / 64, N_PTS / 128), 256, 0, stream>>>(
        h3_16, f1w16, fuse_b1, fu, nullptr, N_PTS, DM, 768, 0);
    ln256<<<N_PTS, 256, 0, stream>>>(fu, fuse_ln_s, fuse_ln_b, nullptr, nullptr, g16, 1);
    wmma_gemm<<<dim3(DM / 64, N_PTS / 128), 256, 0, stream>>>(
        g16, f2w16, fuse_b2, (float*)d_out, nullptr, N_PTS, DM, DM, 0);
}
